// HierarchicalEncoder_30666066493658
// MI455X (gfx1250) — compile-verified, matrix operand order checked by reading, not run on hardware
//
#include <hip/hip_runtime.h>
#include <math.h>

// ---------------------------------------------------------------------------
// CDNA5 (gfx1250) wave32 WMMA types
// ---------------------------------------------------------------------------
typedef __attribute__((ext_vector_type(16))) __bf16 v16bf;
typedef __attribute__((ext_vector_type(8)))  __bf16 v8bf;
typedef __attribute__((ext_vector_type(8)))  float  v8f;

// Map bf16 A/B fragment element e (0..15) + lane-half to K index
// (ISA 7.12.2: 16-bit A 16x32 layout; B mirrors with N in place of M)
__device__ __forceinline__ int kmap(int e, int half) {
    return ((e >> 3) << 4) + (half << 3) + (((e >> 1) & 3) << 1) + (e & 1);
}

__device__ __forceinline__ float gelu_exact(float v) {
    return 0.5f * v * (1.0f + erff(v * 0.70710678118654752f));
}

// Pack 8 consecutive f32 (two float4) into fragment elements [base, base+8)
__device__ __forceinline__ void pack8(v16bf& v, int base, const float4 x, const float4 y) {
    v[base + 0] = (__bf16)x.x; v[base + 1] = (__bf16)x.y;
    v[base + 2] = (__bf16)x.z; v[base + 3] = (__bf16)x.w;
    v[base + 4] = (__bf16)y.x; v[base + 5] = (__bf16)y.y;
    v[base + 6] = (__bf16)y.z; v[base + 7] = (__bf16)y.w;
}
__device__ __forceinline__ void zero8(v16bf& v, int base) {
#pragma unroll
    for (int e = 0; e < 8; ++e) v[base + e] = (__bf16)0.0f;
}
// Load 16 fragment elements from a contiguous f32 k-run starting at p
// (covers e0..7 at p+8*half, e8..15 at p+16+8*half)
__device__ __forceinline__ void load_frag_contig(v16bf& v, const float* p, int half) {
    const float4 x0 = *(const float4*)(p + 8 * half);
    const float4 x1 = *(const float4*)(p + 8 * half + 4);
    const float4 x2 = *(const float4*)(p + 16 + 8 * half);
    const float4 x3 = *(const float4*)(p + 16 + 8 * half + 4);
    pack8(v, 0, x0, x1);
    pack8(v, 8, x2, x3);
}
// Same but source is already bf16 (packed weights): two b128 loads, no cvt
__device__ __forceinline__ v16bf load_frag_bf16(const __bf16* p, int half) {
    const v8bf lo = *(const v8bf*)(p + 8 * half);
    const v8bf hi = *(const v8bf*)(p + 16 + 8 * half);
    return __builtin_shufflevector(lo, hi, 0, 1, 2, 3, 4, 5, 6, 7,
                                   8, 9, 10, 11, 12, 13, 14, 15);
}

// Weight pre-pack: W (K,N) f32 row-major -> Wt (N,K) bf16 row-major
__global__ void pack_wt(const float* __restrict__ W, __bf16* __restrict__ Wt,
                        int K, int N, int total) {
    const int t = blockIdx.x * blockDim.x + threadIdx.x;
    if (t >= total) return;
    const int n = t / K, k = t % K;             // consecutive t -> contiguous writes
    Wt[(size_t)n * K + k] = (__bf16)W[(size_t)k * N + n];
}

// ---------------------------------------------------------------------------
// WMMA GEMM: out = act((A @ W + bias) * g + b) [+ resid]
//   A: (M,K) f32 row-major. Wt: (N,K) bf16 row-major (pre-packed transpose).
//   M % 16 == 0, N % 16 == 0. 8 waves/block, one 16x16 D tile per wave.
// act: 0=none 1=relu 2=gelu
// ---------------------------------------------------------------------------
__global__ void gemm_wmma_bf16(const float* __restrict__ A,
                               const __bf16* __restrict__ Wt,
                               const float* __restrict__ bias,
                               const float* __restrict__ gain,
                               const float* __restrict__ beta,
                               const float* __restrict__ resid,
                               float* __restrict__ out,
                               int M, int N, int K, int act) {
    const int wv   = threadIdx.x >> 5;
    const int lane = threadIdx.x & 31;
    const int tm   = blockIdx.y * 8 + wv;
    const int tn   = blockIdx.x;
    if (tm * 16 >= M) return;           // wave-uniform
    const int m0   = tm * 16;
    const int half = lane >> 4;
    const int l15  = lane & 15;
    const int row  = m0 + l15;          // < M (M multiple of 16)
    const int col  = tn * 16 + l15;     // < N (grid covers N exactly)

    const float*  Arow = A  + (size_t)row * K;
    const __bf16* Wrow = Wt + (size_t)col * K;
    const int K32 = K & ~31;

    v8f acc = {};
    for (int kk = 0; kk < K32; kk += 32) {
        __builtin_prefetch(Arow + (kk + 64 < K ? kk + 64 : K - 1), 0, 1);
        v16bf a;
        load_frag_contig(a, Arow + kk, half);           // 4x b128 + cvt_pk
        const v16bf b = load_frag_bf16(Wrow + kk, half); // 2x b128, no cvt
        acc = __builtin_amdgcn_wmma_f32_16x16x32_bf16(
                  false, a, false, b, (short)0, acc, false, false);
    }
    if (K32 < K) {                      // tail (K % 32 != 0, e.g. K == 3)
        v16bf a, b;
#pragma unroll
        for (int e = 0; e < 16; ++e) {
            const int k  = K32 + kmap(e, half);
            const int kc = (k < K) ? k : (K - 1);       // clamp: loads stay in-bounds
            float av = Arow[kc];
            float bv = (float)Wrow[kc];
            a[e] = (__bf16)((k < K) ? av : 0.0f);       // v_cndmask, no exec branch
            b[e] = (__bf16)((k < K) ? bv : 0.0f);
        }
        acc = __builtin_amdgcn_wmma_f32_16x16x32_bf16(
                  false, a, false, b, (short)0, acc, false, false);
    }
#pragma unroll
    for (int r = 0; r < 8; ++r) {
        const int m = m0 + r + (half << 3);
        const int n = col;
        float v = acc[r];
        if (bias) v += bias[n];
        if (gain) v = v * gain[n] + beta[n];
        if (act == 1) v = fmaxf(v, 0.0f);
        else if (act == 2) v = gelu_exact(v);
        if (resid) v += resid[(size_t)m * N + n];
        out[(size_t)m * N + n] = v;
    }
}

// ---------------------------------------------------------------------------
// Fused radius-masked attention, one wave per (batch, head, 16-query tile).
//   qkv: (B*n, 3*d) rows; q|k|v at col offsets 0|d|2d, head h at +h*HD.
//   S = QK^T*scale - 1e5*[dist>radius]; softmax in LDS; O = P V via WMMA.
// ---------------------------------------------------------------------------
template <int HD>
__global__ void attn_wmma(const float* __restrict__ qkv,
                          const float* __restrict__ cen,
                          float* __restrict__ out,
                          int n, int d, float rad2, float scale) {
    __shared__ float sp[16 * 512];      // 16 query rows x up to 512 keys (32 KB)
    const int lane = threadIdx.x;
    const int qt = blockIdx.x, h = blockIdx.y, b = blockIdx.z;
    const float* base = qkv + (size_t)b * n * 3 * d;
    const float* cb   = cen + (size_t)b * n * 3;
    const int half = lane >> 4, l15 = lane & 15;
    const int qrow = qt * 16 + l15;
    const int d3 = 3 * d;

    const float* qp = base + (size_t)qrow * d3 + h * HD;     // Q row (contig in k)

    // ---- scores: S[16 x n] ----
    for (int jt = 0; jt < n / 16; ++jt) {
        const int key = jt * 16 + l15;
        const float* kp = base + (size_t)key * d3 + d + h * HD; // K row (contig in k)
        v8f acc = {};
#pragma unroll
        for (int kk = 0; kk < HD; kk += 32) {
            v16bf a, bm;
            if (HD - kk >= 32) {                      // full 32-deep slab
                load_frag_contig(a,  qp + kk, half);
                load_frag_contig(bm, kp + kk, half);
            } else {                                  // HD==16: upper half zero
                const float4 q0 = *(const float4*)(qp + kk + 8 * half);
                const float4 q1 = *(const float4*)(qp + kk + 8 * half + 4);
                const float4 k0 = *(const float4*)(kp + kk + 8 * half);
                const float4 k1 = *(const float4*)(kp + kk + 8 * half + 4);
                pack8(a, 0, q0, q1);  zero8(a, 8);
                pack8(bm, 0, k0, k1); zero8(bm, 8);
            }
            acc = __builtin_amdgcn_wmma_f32_16x16x32_bf16(
                      false, a, false, bm, (short)0, acc, false, false);
        }
#pragma unroll
        for (int r = 0; r < 8; ++r) {
            const int qm = qt * 16 + r + (half << 3);
            const int kn = jt * 16 + l15;
            float dx = cb[qm * 3 + 0] - cb[kn * 3 + 0];
            float dy = cb[qm * 3 + 1] - cb[kn * 3 + 1];
            float dz = cb[qm * 3 + 2] - cb[kn * 3 + 2];
            float s = acc[r] * scale;
            if (dx * dx + dy * dy + dz * dz > rad2) s -= 100000.0f;
            sp[(r + (half << 3)) * n + kn] = s;
        }
    }
    __syncthreads();

    // ---- row softmax (lanes 0..15, one row each) ----
    if (lane < 16) {
        float* rp = sp + lane * n;
        float mx = -1e30f;
        for (int j = 0; j < n; ++j) mx = fmaxf(mx, rp[j]);
        float sum = 0.0f;
        for (int j = 0; j < n; ++j) { float e = expf(rp[j] - mx); rp[j] = e; sum += e; }
        float inv = 1.0f / sum;
        for (int j = 0; j < n; ++j) rp[j] *= inv;
    }
    __syncthreads();

    // ---- O = P @ V ----
    const float* prow = sp + l15 * n;                 // P row in LDS (contig in k)
#pragma unroll
    for (int ct = 0; ct < HD / 16; ++ct) {
        v8f acc = {};
        for (int j = 0; j < n; j += 32) {
            v16bf a, bm;
            load_frag_contig(a, prow + j, half);      // ds_load_b128 x4
            const float* vp = base + (size_t)(j + 8 * half) * d3 + 2 * d + h * HD + ct * 16 + l15;
#pragma unroll
            for (int e = 0; e < 8; ++e) bm[e] = (__bf16)vp[(size_t)e * d3];
            const float* vq = vp + (size_t)16 * d3;
#pragma unroll
            for (int e = 0; e < 8; ++e) bm[8 + e] = (__bf16)vq[(size_t)e * d3];
            acc = __builtin_amdgcn_wmma_f32_16x16x32_bf16(
                      false, a, false, bm, (short)0, acc, false, false);
        }
#pragma unroll
        for (int r = 0; r < 8; ++r) {
            const int qm = qt * 16 + r + (half << 3);
            out[((size_t)b * n + qm) * d + h * HD + ct * 16 + l15] = acc[r];
        }
    }
}

// ---------------------------------------------------------------------------
// Farthest point sampling: one block per batch, dist array in LDS.
// ---------------------------------------------------------------------------
__global__ void fps_kernel(const float* __restrict__ xyz, int* __restrict__ outIdx,
                           int n, int np) {
    __shared__ float dist[8192];
    __shared__ float rv[256];
    __shared__ int   ri[256];
    __shared__ int   sfar;
    const int b = blockIdx.x, tid = threadIdx.x;
    const float* p = xyz + (size_t)b * n * 3;
    for (int i = tid; i < n; i += 256) dist[i] = 1e10f;
    if (tid == 0) sfar = 0;
    __syncthreads();
    for (int it = 0; it < np; ++it) {
        const int far = sfar;
        if (tid == 0) outIdx[b * np + it] = far;
        const float cx = p[far * 3], cy = p[far * 3 + 1], cz = p[far * 3 + 2];
        float bm = -1.0f; int bi = 0;
        for (int i = tid; i < n; i += 256) {
            float dx = p[i * 3] - cx, dy = p[i * 3 + 1] - cy, dz = p[i * 3 + 2] - cz;
            float dd = fminf(dist[i], dx * dx + dy * dy + dz * dz);
            dist[i] = dd;
            if (dd > bm) { bm = dd; bi = i; }
        }
        rv[tid] = bm; ri[tid] = bi;
        __syncthreads();
        for (int s = 128; s > 0; s >>= 1) {
            if (tid < s) {
                if (rv[tid + s] > rv[tid] ||
                    (rv[tid + s] == rv[tid] && ri[tid + s] < ri[tid])) {
                    rv[tid] = rv[tid + s]; ri[tid] = ri[tid + s];
                }
            }
            __syncthreads();
        }
        if (tid == 0) sfar = ri[0];
        __syncthreads();
    }
}

// kNN: one thread per (b,center), insertion sort of k nearest (k<=16)
__global__ void knn_kernel(const float* __restrict__ xyz, const float* __restrict__ cen,
                           int* __restrict__ idx, int n, int g, int k, int total) {
    const int t = blockIdx.x * blockDim.x + threadIdx.x;
    if (t >= total) return;
    const int b = t / g;
    const float* p = xyz + (size_t)b * n * 3;
    const float cx = cen[t * 3], cy = cen[t * 3 + 1], cz = cen[t * 3 + 2];
    float bd[16]; int bi[16];
    for (int j = 0; j < k; ++j) { bd[j] = 1e30f; bi[j] = 0; }
    for (int i = 0; i < n; ++i) {
        float dx = p[i * 3] - cx, dy = p[i * 3 + 1] - cy, dz = p[i * 3 + 2] - cz;
        float d2 = dx * dx + dy * dy + dz * dz;
        if (d2 < bd[k - 1]) {
            int j = k - 1;
            while (j > 0 && bd[j - 1] > d2) { bd[j] = bd[j - 1]; bi[j] = bi[j - 1]; --j; }
            bd[j] = d2; bi[j] = i;
        }
    }
    for (int j = 0; j < k; ++j) idx[(size_t)t * k + j] = bi[j];
}

__global__ void gather_xyz(const float* __restrict__ xyz, const int* __restrict__ cidx,
                           float* __restrict__ cen, int n, int g, int total) {
    const int t = blockIdx.x * blockDim.x + threadIdx.x;
    if (t >= total) return;
    const int b = t / g, id = cidx[t];
    for (int c = 0; c < 3; ++c) cen[t * 3 + c] = xyz[((size_t)b * n + id) * 3 + c];
}

__global__ void group_xyz(const float* __restrict__ xyz, const int* __restrict__ idx,
                          const float* __restrict__ cen, float* __restrict__ grp,
                          int n, int g, int k, int total) {
    const int t = blockIdx.x * blockDim.x + threadIdx.x;
    if (t >= total) return;
    const int b = t / (g * k), gi = (t / k) % g, id = idx[t];
    for (int c = 0; c < 3; ++c)
        grp[t * 3 + c] = xyz[((size_t)b * n + id) * 3 + c] - cen[((size_t)b * g + gi) * 3 + c];
}

__global__ void group_feat(const float* __restrict__ x, const int* __restrict__ idx,
                           float* __restrict__ grp, int n, int g, int k, int cin, int total) {
    const int t = blockIdx.x * blockDim.x + threadIdx.x;
    if (t >= total) return;
    const int b = t / (g * k), id = idx[t];
    for (int c = 0; c < cin; ++c)
        grp[(size_t)t * cin + c] = x[((size_t)b * n + id) * cin + c];
}

// max over k within each group: h (g*k, C) -> out (g, C)   (per-batch)
__global__ void maxk_kernel(const float* __restrict__ h, float* __restrict__ out,
                            int g, int k, int C) {
    const int t = blockIdx.x * blockDim.x + threadIdx.x;
    if (t >= g * C) return;
    const int gi = t / C, c = t % C;
    float m = -1e30f;
    for (int j = 0; j < k; ++j) m = fmaxf(m, h[((size_t)gi * k + j) * C + c]);
    out[t] = m;
}

// hcat (rows,2C) = [gmax[row/k] | h2[row]]   (per-batch)
__global__ void concat_kernel(const float* __restrict__ gmax, const float* __restrict__ h2,
                              float* __restrict__ hcat, int rows, int k, int C) {
    const int t = blockIdx.x * blockDim.x + threadIdx.x;
    const int C2 = 2 * C;
    if (t >= rows * C2) return;
    const int row = t / C2, c = t % C2;
    hcat[t] = (c < C) ? gmax[(size_t)(row / k) * C + c]
                      : h2[(size_t)row * C + (c - C)];
}

__global__ void add_kernel(const float* __restrict__ a, const float* __restrict__ b,
                           float* __restrict__ o, int ntot) {
    const int t = blockIdx.x * blockDim.x + threadIdx.x;
    if (t < ntot) o[t] = a[t] + b[t];
}

__global__ void ln_kernel(const float* __restrict__ x, const float* __restrict__ g,
                          const float* __restrict__ bt, float* __restrict__ out,
                          int rows, int d) {
    const int r = blockIdx.x * blockDim.x + threadIdx.x;
    if (r >= rows) return;
    const float* xr = x + (size_t)r * d;
    float m = 0.0f;
    for (int i = 0; i < d; ++i) m += xr[i];
    m /= (float)d;
    float v = 0.0f;
    for (int i = 0; i < d; ++i) { float t = xr[i] - m; v += t * t; }
    v /= (float)d;
    const float inv = rsqrtf(v + 1e-5f);
    float* o = out + (size_t)r * d;
    for (int i = 0; i < d; ++i) o[i] = (xr[i] - m) * inv * g[i] + bt[i];
}

// ---------------------------------------------------------------------------
// Host orchestration
// ---------------------------------------------------------------------------
namespace {
struct Lin { const float* w; const float* b; };
struct Aff { const float* g; const float* b; };
struct Blk { Lin fc1, fc2; Aff ln1, ln2; Lin proj; const float* qkvw; };

inline void gemm(hipStream_t s, const float* A, const __bf16* Wt, const float* bias,
                 const float* gain, const float* beta, const float* resid,
                 float* out, int M, int N, int K, int act) {
    dim3 grid((N + 15) / 16, (M + 127) / 128);
    gemm_wmma_bf16<<<grid, 256, 0, s>>>(A, Wt, bias, gain, beta, resid, out, M, N, K, act);
}
inline int cdiv(int a, int b) { return (a + b - 1) / b; }
} // namespace

extern "C" void kernel_launch(void* const* d_in, const int* in_sizes, int n_in,
                              void* d_out, int out_size, void* d_ws, size_t ws_size,
                              hipStream_t stream) {
    (void)in_sizes; (void)n_in; (void)out_size; (void)ws_size;
    static const int B = 32, NPTS = 8192, HEADS = 6, DEP = 5;
    static const int GSZ[3] = {16, 8, 8};
    static const int NG[3]  = {512, 256, 64};
    static const int DIM[3] = {96, 192, 384};
    static const float RAD2[3] = {0.32f * 0.32f, 0.64f * 0.64f, 1.28f * 1.28f};

    const float* pts = (const float*)d_in[0];

    // ---- unflatten params (jax tree order: dict keys sorted; None dropped) ----
    int p = 1;
    auto nxt = [&]() { return (const float*)d_in[p++]; };
    Blk blocks[3][DEP];
    for (int i = 0; i < 3; ++i)
        for (int j = 0; j < DEP; ++j) {
            Blk& bk = blocks[i][j];
            bk.fc1.b = nxt(); bk.fc1.w = nxt();
            bk.fc2.b = nxt(); bk.fc2.w = nxt();
            bk.ln1.b = nxt(); bk.ln1.g = nxt();
            bk.ln2.b = nxt(); bk.ln2.g = nxt();
            bk.proj.b = nxt(); bk.proj.w = nxt();
            bk.qkvw = nxt();                    // qkv bias is None -> dropped
        }
    Aff norm; norm.b = nxt(); norm.g = nxt();
    Lin pos1[3], pos2[3];
    for (int i = 0; i < 3; ++i) {
        pos1[i].b = nxt(); pos1[i].w = nxt();
        pos2[i].b = nxt(); pos2[i].w = nxt();
    }
    struct TE { Aff bn1, bn2; Lin c1a, c1b, c2a, c2b; } te[3];
    for (int i = 0; i < 3; ++i) {
        te[i].bn1.b = nxt(); te[i].bn1.g = nxt();
        te[i].bn2.b = nxt(); te[i].bn2.g = nxt();
        te[i].c1a.b = nxt(); te[i].c1a.w = nxt();
        te[i].c1b.b = nxt(); te[i].c1b.w = nxt();
        te[i].c2a.b = nxt(); te[i].c2a.w = nxt();
        te[i].c2b.b = nxt(); te[i].c2b.w = nxt();
    }

    // ---- workspace bump allocator ----
    char* ws = (char*)d_ws;
    size_t off = 0;
    auto allocB = [&](size_t bytes) -> void* {
        off = (off + 255) & ~(size_t)255;
        void* r = ws + off;
        off += bytes;
        return r;
    };
    auto allocF = [&](size_t elems) { return (float*)allocB(elems * sizeof(float)); };
    auto allocI = [&](size_t elems) { return (int*)allocB(elems * sizeof(int)); };

    // pack weight W (K,N) f32 -> (N,K) bf16 on stream; bump-allocated
    auto packw = [&](const float* w, int K, int N) -> const __bf16* {
        __bf16* wt = (__bf16*)allocB((size_t)K * N * sizeof(__bf16));
        const int tot = K * N;
        pack_wt<<<cdiv(tot, 256), 256, 0, stream>>>(w, wt, K, N, tot);
        return wt;
    };

    // ---- pre-pack all GEMM weights to transposed bf16 ----
    struct PBlk { const __bf16 *qkv, *proj, *fc1, *fc2; } pblk[3][DEP];
    const __bf16 *ppos1[3], *ppos2[3];
    struct PTE { const __bf16 *c1a, *c1b, *c2a, *c2b; } pte[3];
    for (int i = 0; i < 3; ++i) {
        const int d = DIM[i];
        const int cin = (i == 0) ? 3 : DIM[i - 1];
        const int C1 = (i == 0) ? 128 : cin;
        const int C2 = (i == 0) ? 256 : cin;
        const int C3 = (i == 0) ? 512 : d;
        for (int j = 0; j < DEP; ++j) {
            pblk[i][j].qkv  = packw(blocks[i][j].qkvw,   d,     3 * d);
            pblk[i][j].proj = packw(blocks[i][j].proj.w, d,     d);
            pblk[i][j].fc1  = packw(blocks[i][j].fc1.w,  d,     4 * d);
            pblk[i][j].fc2  = packw(blocks[i][j].fc2.w,  4 * d, d);
        }
        ppos1[i] = packw(pos1[i].w, 3, d);
        ppos2[i] = packw(pos2[i].w, d, d);
        pte[i].c1a = packw(te[i].c1a.w, cin,    C1);
        pte[i].c1b = packw(te[i].c1b.w, C1,     C2);
        pte[i].c2a = packw(te[i].c2a.w, 2 * C2, C3);
        pte[i].c2b = packw(te[i].c2b.w, C3,     d);
    }

    int*   cidx    = allocI((size_t)B * 512);
    float* cen0    = allocF((size_t)B * 512 * 3);
    float* cen1    = allocF((size_t)B * 256 * 3);
    float* cen2    = allocF((size_t)B * 64 * 3);
    float* cenArr[3] = {cen0, cen1, cen2};
    int*   knnidx  = allocI((size_t)B * 512 * 16);
    float* grp     = allocF((size_t)B * 256 * 8 * 96);        // max across stages
    float* h1      = allocF((size_t)8192 * 192);
    float* h2      = allocF((size_t)8192 * 256);
    float* gmax    = allocF((size_t)512 * 384);
    float* hcat    = allocF((size_t)8192 * 512);
    float* h3      = allocF((size_t)8192 * 512);
    float* h4      = allocF((size_t)8192 * 384);
    float* hBuf    = allocF((size_t)B * 512 * 96);            // tokens
    float* posBuf  = allocF((size_t)B * 512 * 96);
    float* bufX    = allocF((size_t)B * 512 * 96);
    float* bufX2   = allocF((size_t)B * 512 * 96);
    float* bufA    = allocF((size_t)B * 512 * 96);
    float* bufL    = allocF((size_t)B * 512 * 96);
    float* qkvBuf  = allocF((size_t)B * 512 * 288);
    float* bufH    = allocF((size_t)B * 512 * 384);           // mlp hidden / pos tmp

    for (int i = 0; i < 3; ++i) {
        const int g = NG[i], k = GSZ[i], d = DIM[i];
        const int nsrc = (i == 0) ? NPTS : NG[i - 1];
        const float* xyz = (i == 0) ? pts : cenArr[i - 1];
        const int cin = (i == 0) ? 3 : DIM[i - 1];
        const int C1 = (i == 0) ? 128 : cin;
        const int C2 = (i == 0) ? 256 : cin;
        const int C3 = (i == 0) ? 512 : d;
        const int hd = d / HEADS;
        const float scale = rsqrtf((float)hd);
        const int rows = B * g;

        // 1) FPS + centers + kNN + grouping
        fps_kernel<<<B, 256, 0, stream>>>(xyz, cidx, nsrc, g);
        gather_xyz<<<cdiv(B * g, 256), 256, 0, stream>>>(xyz, cidx, cenArr[i], nsrc, g, B * g);
        knn_kernel<<<cdiv(B * g, 256), 256, 0, stream>>>(xyz, cenArr[i], knnidx, nsrc, g, k, B * g);
        const int totGK = B * g * k;
        if (i == 0)
            group_xyz<<<cdiv(totGK, 256), 256, 0, stream>>>(xyz, knnidx, cenArr[0], grp,
                                                            nsrc, g, k, totGK);
        else
            group_feat<<<cdiv(totGK, 256), 256, 0, stream>>>(hBuf, knnidx, grp,
                                                             nsrc, g, k, cin, totGK);

        // 2) token embed (chunked per batch to bound workspace)
        for (int b = 0; b < B; ++b) {
            const float* gb = grp + (size_t)b * g * k * cin;
            const int rb = g * k;
            gemm(stream, gb, pte[i].c1a, te[i].c1a.b, te[i].bn1.g, te[i].bn1.b,
                 nullptr, h1, rb, C1, cin, /*relu*/1);
            gemm(stream, h1, pte[i].c1b, te[i].c1b.b, nullptr, nullptr,
                 nullptr, h2, rb, C2, C1, 0);
            maxk_kernel<<<cdiv(g * C2, 256), 256, 0, stream>>>(h2, gmax, g, k, C2);
            concat_kernel<<<cdiv(rb * 2 * C2, 256), 256, 0, stream>>>(gmax, h2, hcat, rb, k, C2);
            gemm(stream, hcat, pte[i].c2a, te[i].c2a.b, te[i].bn2.g, te[i].bn2.b,
                 nullptr, h3, rb, C3, 2 * C2, /*relu*/1);
            gemm(stream, h3, pte[i].c2b, te[i].c2b.b, nullptr, nullptr,
                 nullptr, h4, rb, d, C3, 0);
            maxk_kernel<<<cdiv(g * d, 256), 256, 0, stream>>>(h4, hBuf + (size_t)b * g * d, g, k, d);
        }

        // 3) positional embedding: gelu(centers@l1) @ l2
        gemm(stream, cenArr[i], ppos1[i], pos1[i].b, nullptr, nullptr,
             nullptr, bufH, rows, d, 3, /*gelu*/2);
        gemm(stream, bufH, ppos2[i], pos2[i].b, nullptr, nullptr,
             nullptr, posBuf, rows, d, d, 0);

        // 4) transformer blocks
        for (int j = 0; j < DEP; ++j) {
            const Blk& bk = blocks[i][j];
            const PBlk& pb = pblk[i][j];
            add_kernel<<<cdiv(rows * d, 256), 256, 0, stream>>>(hBuf, posBuf, bufX, rows * d);
            ln_kernel<<<cdiv(rows, 128), 128, 0, stream>>>(bufX, bk.ln1.g, bk.ln1.b, bufL, rows, d);
            gemm(stream, bufL, pb.qkv, nullptr, nullptr, nullptr,
                 nullptr, qkvBuf, rows, 3 * d, d, 0);
            dim3 agrid(g / 16, HEADS, B);
            if (hd == 16)
                attn_wmma<16><<<agrid, 32, 0, stream>>>(qkvBuf, cenArr[i], bufA, g, d, RAD2[i], scale);
            else if (hd == 32)
                attn_wmma<32><<<agrid, 32, 0, stream>>>(qkvBuf, cenArr[i], bufA, g, d, RAD2[i], scale);
            else
                attn_wmma<64><<<agrid, 32, 0, stream>>>(qkvBuf, cenArr[i], bufA, g, d, RAD2[i], scale);
            gemm(stream, bufA, pb.proj, bk.proj.b, nullptr, nullptr,
                 /*resid*/bufX, bufX2, rows, d, d, 0);
            ln_kernel<<<cdiv(rows, 128), 128, 0, stream>>>(bufX2, bk.ln2.g, bk.ln2.b, bufL, rows, d);
            gemm(stream, bufL, pb.fc1, bk.fc1.b, nullptr, nullptr,
                 nullptr, bufH, rows, 4 * d, d, /*gelu*/2);
            gemm(stream, bufH, pb.fc2, bk.fc2.b, nullptr, nullptr,
                 /*resid*/bufX2, hBuf, rows, d, 4 * d, 0);
        }
    }

    // 5) final layernorm -> d_out (B, 64, 384)
    ln_kernel<<<cdiv(B * NG[2], 128), 128, 0, stream>>>(
        hBuf, norm.g, norm.b, (float*)d_out, B * NG[2], DIM[2]);
}
